// FeatureExtractorV2_84413287236435
// MI455X (gfx1250) — matrix operand order repaired and checked by reading, not compiled, not run on hardware
//
#include <hip/hip_runtime.h>
#include <stdint.h>

#define N_BRANCH 32
#define T_SEQ    2048
#define HID      256
#define G4H      1024   // 4*H
#define EDIM     512    // 2*H
#define DMAX     12
#define NHEAD    4
#define DHEAD    128    // EDIM / NHEAD

typedef __attribute__((ext_vector_type(16))) __bf16 v16bf;
typedef __attribute__((ext_vector_type(8)))  float  v8f;

__device__ __forceinline__ float sigf(float x) { return 1.0f / (1.0f + __expf(-x)); }

// ---------------------------------------------------------------------------
// 1) Pack w_hh (fp32 [n][4H][H]) -> bf16-pair, transposed [dir][n][k2][g]
//    so the scan kernel's per-step weight stream is coalesced across threads.
// ---------------------------------------------------------------------------
__global__ void pack_whh_kernel(const float* __restrict__ whf,
                                const float* __restrict__ whb,
                                uint32_t* __restrict__ wpack) {
  int idx = blockIdx.x * blockDim.x + threadIdx.x;   // [0, 2*32*128*1024)
  int g   = idx & 1023;
  int k2  = (idx >> 10) & 127;
  int n   = (idx >> 17) & 31;
  int dir = idx >> 22;
  const float* src = (dir ? whb : whf) + (size_t)(n * G4H + g) * HID + 2 * k2;
  uint32_t b0 = __float_as_uint(src[0]);
  uint32_t b1 = __float_as_uint(src[1]);
  b0 += 0x7fffu + ((b0 >> 16) & 1u);                 // RNE to bf16
  b1 += 0x7fffu + ((b1 >> 16) & 1u);
  wpack[idx] = (b0 >> 16) | ((b1 >> 16) << 16);
}

// ---------------------------------------------------------------------------
// 2) Bidirectional LSTM scan. One 1024-thread WG per (branch, direction).
//    Thread g owns gate g: 12-MAC input projection from registers + 256-MAC
//    recurrent dot (bf16 weights from L2, h broadcast from LDS as float2).
// ---------------------------------------------------------------------------
__global__ void __launch_bounds__(1024)
lstm_scan_kernel(const float* __restrict__ x,
                 const float* __restrict__ wih_f, const float* __restrict__ b_f,
                 const float* __restrict__ wih_b, const float* __restrict__ b_b,
                 const uint32_t* __restrict__ wpack,
                 float* __restrict__ embs) {
  const int blk = blockIdx.x;
  const int dir = blk >> 5;
  const int n   = blk & 31;
  const int tg  = threadIdx.x;

  __shared__ __align__(16) float lds_h[HID];
  __shared__ float lds_gates[G4H];
  __shared__ float lds_x[DMAX];

  const float* wih = (dir ? wih_b : wih_f) + (size_t)(n * G4H + tg) * DMAX;
  float wreg[DMAX];
  #pragma unroll
  for (int j = 0; j < DMAX; ++j) wreg[j] = wih[j];
  const float bias = (dir ? b_b : b_f)[n * G4H + tg];
  const uint32_t* wp = wpack + (size_t)(dir * N_BRANCH + n) * 128 * G4H;
  const float* xb = x + (size_t)n * T_SEQ * DMAX;

  float creg = 0.0f;
  if (tg < HID) lds_h[tg] = 0.0f;
  __syncthreads();

  for (int step = 0; step < T_SEQ; ++step) {
    const int t = dir ? (T_SEQ - 1 - step) : step;
    if (tg < DMAX) lds_x[tg] = xb[t * DMAX + tg];
    __syncthreads();

    float acc = bias;
    #pragma unroll
    for (int j = 0; j < DMAX; ++j) acc = fmaf(wreg[j], lds_x[j], acc);

    const float2* h2p = reinterpret_cast<const float2*>(lds_h);
    #pragma unroll 4
    for (int k2 = 0; k2 < 128; ++k2) {
      uint32_t w2 = wp[(size_t)k2 * G4H + tg];
      float2 h2 = h2p[k2];
      acc = fmaf(__uint_as_float(w2 << 16),         h2.x, acc);
      acc = fmaf(__uint_as_float(w2 & 0xffff0000u), h2.y, acc);
    }
    lds_gates[tg] = acc;
    __syncthreads();

    if (tg < HID) {                       // torch gate order: i, f, g, o
      float ig = sigf(lds_gates[tg]);
      float fg = sigf(lds_gates[HID + tg]);
      float gg = tanhf(lds_gates[2 * HID + tg]);
      float og = sigf(lds_gates[3 * HID + tg]);
      creg = fg * creg + ig * gg;
      float h = og * tanhf(creg);
      lds_h[tg] = h;
      embs[((size_t)n * T_SEQ + t) * EDIM + dir * HID + tg] = h;
    }
    __syncthreads();
  }
}

// ---------------------------------------------------------------------------
// 3) C = A @ W^T + bias_row(batch), bf16 WMMA, fp32 accumulate.
//    Each wave computes a (16*MT)x(16*FT) C strip: per k-step MT A-frags +
//    FT B-frags feed MT*FT v_wmma_f32_16x16x32_bf16.
//    Fat strip (2x4) for batched GEMMs; lean strip (1x2) for single-batch
//    GEMMs where wave count, not per-wave efficiency, limits occupancy.
//    Fragments per CDNA5 16-bit A(16x32) / B(32x16) lane layouts.
//    bias indexed as bias[bb*sBias + f] (per-batch bias rows fold type_emb).
// ---------------------------------------------------------------------------
template <int MT, int FT>
__global__ void __launch_bounds__(256)
wmma_gemm_kernel(const float* __restrict__ A, long long sA, int lda,
                 const float* __restrict__ W,
                 const float* __restrict__ bias, long long sBias,
                 float* __restrict__ C, long long sC, int ldc,
                 int M, int K, int F) {
  const int bb = blockIdx.y;
  const float* Ab = A + (size_t)bb * sA;
  const float* Bb = bias + (size_t)bb * sBias;
  float*       Cb = C + (size_t)bb * sC;

  const int wid  = (blockIdx.x * blockDim.x + threadIdx.x) >> 5;
  const int lane = threadIdx.x & 31;
  const int strips_m = M / (16 * MT);
  const int strips_f = F / (16 * FT);
  if (wid >= strips_m * strips_f) return;
  const int sm = wid % strips_m;
  const int sf = wid / strips_m;

  const int mrow = lane & 15;             // A row within tile / B col within tile
  const int half = lane >> 4;

  const float* Arow[MT];
  #pragma unroll
  for (int mi = 0; mi < MT; ++mi)
    Arow[mi] = Ab + (size_t)(sm * 16 * MT + mi * 16 + mrow) * lda;
  const float* Wrow[FT];
  #pragma unroll
  for (int fi = 0; fi < FT; ++fi)
    Wrow[fi] = W + (size_t)(sf * 16 * FT + fi * 16 + mrow) * K;

  v8f acc[MT][FT];
  #pragma unroll
  for (int mi = 0; mi < MT; ++mi)
    #pragma unroll
    for (int fi = 0; fi < FT; ++fi)
      acc[mi][fi] = {};

  for (int k0 = 0; k0 < K; k0 += 32) {
    v16bf af[MT], bf[FT];
    // A 16x32: lanes 0-15 hold K = 0..7 / 16..23, lanes 16-31 hold 8..15 / 24..31
    #pragma unroll
    for (int mi = 0; mi < MT; ++mi) {
      #pragma unroll
      for (int j = 0; j < 8; ++j) {
        int k = k0 + ((j < 4) ? (half * 8 + 2 * j) : (16 + half * 8 + 2 * (j - 4)));
        float2 p = *reinterpret_cast<const float2*>(Arow[mi] + k);
        af[mi][2 * j]     = (__bf16)p.x;
        af[mi][2 * j + 1] = (__bf16)p.y;
      }
    }
    // B 32x16: lanes 0-15 hold K = 0..7 / 16..23, lanes 16-31 hold 8..15 / 24..31
    #pragma unroll
    for (int fi = 0; fi < FT; ++fi) {
      #pragma unroll
      for (int v = 0; v < 8; ++v) {
        int k = k0 + ((v < 4) ? (half * 8 + 2 * v) : (16 + half * 8 + 2 * (v - 4)));
        float2 p = *reinterpret_cast<const float2*>(Wrow[fi] + k);
        bf[fi][2 * v]     = (__bf16)p.x;
        bf[fi][2 * v + 1] = (__bf16)p.y;
      }
    }
    #pragma unroll
    for (int mi = 0; mi < MT; ++mi)
      #pragma unroll
      for (int fi = 0; fi < FT; ++fi)
        acc[mi][fi] = __builtin_amdgcn_wmma_f32_16x16x32_bf16(
            false, af[mi], false, bf[fi], (short)0, acc[mi][fi], false, false);
  }

  #pragma unroll
  for (int fi = 0; fi < FT; ++fi) {
    const int fcol = sf * 16 * FT + fi * 16 + mrow;
    const float bsv = Bb[fcol];
    #pragma unroll
    for (int mi = 0; mi < MT; ++mi) {
      #pragma unroll
      for (int r = 0; r < 8; ++r) {       // C: VGPR r -> M = r + half*8, N = lane&15
        int m = sm * 16 * MT + mi * 16 + half * 8 + r;
        Cb[(size_t)m * ldc + fcol] = acc[mi][fi][r] + bsv;
      }
    }
  }
}

// ---------------------------------------------------------------------------
// 4) Per-time-step attention: wave h = one head, lane s = one branch slot.
//    Wave softmax via __shfl_xor (wave32); also emits head-averaged weights.
// ---------------------------------------------------------------------------
__global__ void __launch_bounds__(128)
attn_kernel(const float* __restrict__ qbuf,
            const float* __restrict__ kbuf,
            const float* __restrict__ vbuf,
            float* __restrict__ ctxbuf,
            float* __restrict__ attn_out) {
  const int b = blockIdx.x;
  const int h = threadIdx.x >> 5;
  const int s = threadIdx.x & 31;
  __shared__ float lds_w[NHEAD][32];

  const float4* q4 = reinterpret_cast<const float4*>(qbuf + (size_t)b * EDIM + h * DHEAD);
  const float4* k4 = reinterpret_cast<const float4*>(kbuf + ((size_t)s * T_SEQ + b) * EDIM + h * DHEAD);
  float sc = 0.0f;
  #pragma unroll 8
  for (int j = 0; j < DHEAD / 4; ++j) {
    float4 qa = q4[j], kb = k4[j];
    sc += qa.x * kb.x + qa.y * kb.y + qa.z * kb.z + qa.w * kb.w;
  }
  sc *= 0.08838834764831845f;           // 1/sqrt(128)

  float mx = sc;
  #pragma unroll
  for (int o = 16; o > 0; o >>= 1) mx = fmaxf(mx, __shfl_xor(mx, o, 32));
  float p = __expf(sc - mx);
  float sm = p;
  #pragma unroll
  for (int o = 16; o > 0; o >>= 1) sm += __shfl_xor(sm, o, 32);
  float wgt = p / sm;
  lds_w[h][s] = wgt;
  __syncthreads();

  float4 av = {0.f, 0.f, 0.f, 0.f};     // lane covers d = s*4 .. s*4+3 of head h
  #pragma unroll 4
  for (int ss = 0; ss < N_BRANCH; ++ss) {
    float wv = lds_w[h][ss];
    float4 vv = *reinterpret_cast<const float4*>(
        vbuf + ((size_t)ss * T_SEQ + b) * EDIM + h * DHEAD + s * 4);
    av.x = fmaf(wv, vv.x, av.x); av.y = fmaf(wv, vv.y, av.y);
    av.z = fmaf(wv, vv.z, av.z); av.w = fmaf(wv, vv.w, av.w);
  }
  *reinterpret_cast<float4*>(ctxbuf + (size_t)b * EDIM + h * DHEAD + s * 4) = av;

  if (threadIdx.x < 32) {
    float a = (lds_w[0][s] + lds_w[1][s] + lds_w[2][s] + lds_w[3][s]) * 0.25f;
    attn_out[(size_t)b * N_BRANCH + s] = a;
  }
}

// ---------------------------------------------------------------------------
extern "C" void kernel_launch(void* const* d_in, const int* in_sizes, int n_in,
                              void* d_out, int out_size, void* d_ws, size_t ws_size,
                              hipStream_t stream) {
  (void)in_sizes; (void)n_in; (void)out_size; (void)ws_size;
  const float* x     = (const float*)d_in[0];
  const float* wih_f = (const float*)d_in[1];
  const float* whh_f = (const float*)d_in[2];
  const float* b_f   = (const float*)d_in[3];
  const float* wih_b = (const float*)d_in[4];
  const float* whh_b = (const float*)d_in[5];
  const float* b_b   = (const float*)d_in[6];
  const float* temb  = (const float*)d_in[7];
  const float* wq    = (const float*)d_in[8];
  const float* bq    = (const float*)d_in[9];
  const float* wk    = (const float*)d_in[10];
  const float* bk    = (const float*)d_in[11];
  const float* wv    = (const float*)d_in[12];
  const float* bv    = (const float*)d_in[13];
  const float* wo    = (const float*)d_in[14];
  const float* bo    = (const float*)d_in[15];

  float* out  = (float*)d_out;                       // (2048, 512)
  float* attn = out + (size_t)T_SEQ * EDIM;          // (2048, 32)

  char* ws = (char*)d_ws;
  size_t off = 0;
  uint32_t* wpack = (uint32_t*)(ws + off); off += (size_t)2 * N_BRANCH * 128 * G4H * 4;
  float* embs  = (float*)(ws + off); off += (size_t)N_BRANCH * T_SEQ * EDIM * 4;
  float* qbuf  = (float*)(ws + off); off += (size_t)T_SEQ * EDIM * 4;
  float* kbuf  = (float*)(ws + off); off += (size_t)N_BRANCH * T_SEQ * EDIM * 4;
  float* vbuf  = (float*)(ws + off); off += (size_t)N_BRANCH * T_SEQ * EDIM * 4;
  float* ctxb  = (float*)(ws + off); off += (size_t)T_SEQ * EDIM * 4;
  float* kbias = (float*)(ws + off); off += (size_t)N_BRANCH * EDIM * 4;

  // 1) weight pack (bf16, transposed for coalesced per-step streaming)
  {
    int total = 2 * N_BRANCH * 128 * G4H;
    pack_whh_kernel<<<total / 256, 256, 0, stream>>>(whh_f, whh_b, wpack);
  }
  // 2) bidirectional LSTM scan -> embs (n, t, [fwd|bwd])
  lstm_scan_kernel<<<64, 1024, 0, stream>>>(x, wih_f, b_f, wih_b, b_b, wpack, embs);

  // 3) projections with bf16 WMMA (fp32 accumulate)
  {
    const long long sE = (long long)T_SEQ * EDIM;
    dim3 blk(256);
    // 3a) fold type_emb into per-branch K bias: kbias[n] = bk + temb[n] @ wk^T
    //     lean 1x2 strips: (32/16)*(512/32) = 32 waves -> 4 blocks
    wmma_gemm_kernel<1, 2><<<dim3(4, 1), blk, 0, stream>>>(
        temb, 0, EDIM, wk, bk, 0, kbias, 0, EDIM, N_BRANCH, EDIM, EDIM);
    // 3b) Q: single batch -> lean 1x2 strips: 128*16 = 2048 waves -> 256 blocks
    wmma_gemm_kernel<1, 2><<<dim3(256, 1), blk, 0, stream>>>(
        embs, 0, EDIM, wq, bq, 0, qbuf, 0, EDIM, T_SEQ, EDIM, EDIM);
    // 3c) K/V: 32 batches -> fat 2x4 strips: 64*8 = 512 waves x 32 batches
    wmma_gemm_kernel<2, 4><<<dim3(64, N_BRANCH), blk, 0, stream>>>(
        embs, sE, EDIM, wk, kbias, EDIM, kbuf, sE, EDIM, T_SEQ, EDIM, EDIM);
    wmma_gemm_kernel<2, 4><<<dim3(64, N_BRANCH), blk, 0, stream>>>(
        embs, sE, EDIM, wv, bv, 0, vbuf, sE, EDIM, T_SEQ, EDIM, EDIM);
  }
  // 4) scores + softmax + context + head-averaged attention weights
  attn_kernel<<<T_SEQ, 128, 0, stream>>>(qbuf, kbuf, vbuf, ctxb, attn);

  // 5) output projection -> d_out (lean strips for occupancy)
  wmma_gemm_kernel<1, 2><<<dim3(256, 1), dim3(256), 0, stream>>>(
      ctxb, 0, EDIM, wo, bo, 0, out, 0, EDIM, T_SEQ, EDIM, EDIM);
}
